// OptimizedEnhancedGNN_51196010169103
// MI455X (gfx1250) — matrix operand order repaired and checked by reading, not compiled, not run on hardware
//
#include <hip/hip_runtime.h>
#include <math.h>

typedef __attribute__((ext_vector_type(16))) _Float16 v16h;
typedef __attribute__((ext_vector_type(8)))  _Float16 v8h;
typedef __attribute__((ext_vector_type(8)))  float    v8f;

#define NN    100000
#define EE    1600000
#define GG    64
#define NPAD  100032          /* 64 * 1563 */
#define ET    (EE + NN)       /* edges + self loops */
#define D1    128
#define HEADS 4
#define HID   32
#define OUTD  10
#define LRELU 0.2f
#define BN_EPS 1e-5f
#define ENC_NEG_INF 0x007FFFFFu

__device__ __forceinline__ unsigned fenc(float f) {
  unsigned u = __float_as_uint(f);
  return (u & 0x80000000u) ? ~u : (u | 0x80000000u);
}
__device__ __forceinline__ float fdec(unsigned e) {
  return __uint_as_float((e & 0x80000000u) ? (e & 0x7FFFFFFFu) : ~e);
}
__device__ __forceinline__ void atomAddF(float* p, float v) {
  __hip_atomic_fetch_add(p, v, __ATOMIC_RELAXED, __HIP_MEMORY_SCOPE_AGENT);
}

/* ---------------- generic fills ---------------- */
__global__ void k_fill_f32(float* p, long long n, float v) {
  long long i = (long long)blockIdx.x * blockDim.x + threadIdx.x;
  if (i < n) p[i] = v;
}
__global__ void k_fill_u32(unsigned* p, long long n, unsigned v) {
  long long i = (long long)blockIdx.x * blockDim.x + threadIdx.x;
  if (i < n) p[i] = v;
}

/* ---------------- positions ---------------- */
__global__ void k_count(const int* __restrict__ batch, int* __restrict__ counts) {
  int i = blockIdx.x * blockDim.x + threadIdx.x;
  if (i < NN) atomicAdd(&counts[batch[i]], 1);
}
__global__ void k_scan(const int* __restrict__ counts, int* __restrict__ starts) {
  if (threadIdx.x == 0 && blockIdx.x == 0) {
    int acc = 0;
    for (int g = 0; g < GG; ++g) { starts[g] = acc; acc += counts[g]; }
  }
}

/* build padded f16 input  h0 = [x | pos@pos_w + pos_b | 0pad]  (stride 160) */
__global__ void k_build_h0(const float* __restrict__ x, const int* __restrict__ batch,
                           const int* __restrict__ counts, const int* __restrict__ starts,
                           const float* __restrict__ pos_w, const float* __restrict__ pos_b,
                           _Float16* __restrict__ h0) {
  int n = blockIdx.x * blockDim.x + threadIdx.x;
  if (n >= NPAD) return;
  _Float16* row = h0 + (size_t)n * 160;
  if (n >= NN) { for (int k = 0; k < 160; ++k) row[k] = (_Float16)0.f; return; }
  const float* xr = x + (size_t)n * 128;
  for (int c = 0; c < 128; ++c) row[c] = (_Float16)xr[c];
  int b = batch[n];
  int cnt = counts[b];
  int local = n - starts[b];
  int g = (int)ceilf(sqrtf((float)cnt)); if (g < 1) g = 1;
  float den = (float)((g - 1) > 1 ? (g - 1) : 1);
  float pr = (float)(local / g) / den;
  float pc = (float)(local % g) / den;
  for (int j = 0; j < 16; ++j)
    row[128 + j] = (_Float16)(pr * pos_w[j] + pc * pos_w[16 + j] + pos_b[j]);
  for (int j = 144; j < 160; ++j) row[j] = (_Float16)0.f;
}

/* ---------------- pack W (K x 128 f32, row major) into WMMA B-fragment layout ----------------
 * index = kt*4096 + nt*512 + lane*16 + i ;  k = kt*32 + (lane>=16 ? 16:0) + i ; n = nt*16 + (lane&15) */
__global__ void k_pack_w(const float* __restrict__ W, int K, int KT, _Float16* __restrict__ Wp) {
  int j = blockIdx.x * blockDim.x + threadIdx.x;
  int total = KT * 4096;
  if (j >= total) return;
  int i    = j & 15;
  int lane = (j >> 4) & 31;
  int nt   = (j >> 9) & 7;
  int kt   = j >> 12;
  int k = kt * 32 + ((lane >= 16) ? 16 : 0) + i;
  int n = nt * 16 + (lane & 15);
  float v = (k < K) ? W[(size_t)k * 128 + n] : 0.f;
  Wp[j] = (_Float16)v;
}

/* ---------------- WMMA GEMM:  C[NPAD x 128] = A[NPAD x lda(f16)] @ W  (+bias) ----------------
 * 4 waves / block, one 16x128 output strip per wave, K loop of 32 per step. */
__global__ __launch_bounds__(128) void k_gemm_wmma(const _Float16* __restrict__ A, int lda, int KT,
                                                   const _Float16* __restrict__ Wp,
                                                   const float* __restrict__ bias,
                                                   float* __restrict__ C) {
  __shared__ __align__(32) _Float16 sW[5 * 4096];  /* up to 40 KB of packed weights */
  int tid  = threadIdx.x;
  int wave = tid >> 5;
  int lane = tid & 31;
  int total = KT * 4096;
  for (int i = tid * 8; i < total; i += 128 * 8)
    *(v8h*)(sW + i) = *(const v8h*)(Wp + i);
  __syncthreads();

  int tileBase = (blockIdx.x * 4 + wave) * 16;
  const _Float16* arow = A + (size_t)(tileBase + (lane & 15)) * lda;
  int koff = (lane >> 4) << 4;  /* +16 for upper half-wave */

  v8f acc[8];
#pragma unroll
  for (int nt = 0; nt < 8; ++nt) acc[nt] = (v8f)(0.f);

  for (int kt = 0; kt < KT; ++kt) {
    int kb = kt * 32 + koff;
    v8h alo = *(const v8h*)(arow + kb);
    v8h ahi = *(const v8h*)(arow + kb + 8);
    v16h a = __builtin_shufflevector(alo, ahi, 0,1,2,3,4,5,6,7,8,9,10,11,12,13,14,15);
    if (kt + 1 < KT) __builtin_prefetch((const void*)(arow + kb + 32), 0, 0);
    const _Float16* wbase = sW + (size_t)kt * 4096 + lane * 16;
#pragma unroll
    for (int nt = 0; nt < 8; ++nt) {
      v16h b = *(const v16h*)(wbase + nt * 512);
      acc[nt] = __builtin_amdgcn_wmma_f32_16x16x32_f16(
          /*neg_a=*/false, a, /*neg_b=*/false, b,
          /*c_mod=*/(short)0, acc[nt], /*reuse_a=*/false, /*reuse_b=*/false);
    }
  }

  int rbase = tileBase + ((lane >> 4) << 3);
  int cl = lane & 15;
#pragma unroll
  for (int nt = 0; nt < 8; ++nt) {
    int cc = nt * 16 + cl;
    float bv = bias ? bias[cc] : 0.f;
#pragma unroll
    for (int j = 0; j < 8; ++j)
      C[(size_t)(rbase + j) * 128 + cc] = acc[nt][j] + bv;
  }
}

/* ---------------- per-node attention scalars ---------------- */
__global__ void k_scores(const float* __restrict__ H, const float* __restrict__ asrc,
                         const float* __restrict__ adst,
                         float* __restrict__ es, float* __restrict__ ed) {
  int idx = blockIdx.x * blockDim.x + threadIdx.x;
  if (idx >= NN * HEADS) return;
  int n = idx >> 2, h = idx & 3;
  const float* hp = H + (size_t)n * 128 + h * HID;
  const float* as = asrc + h * HID;
  const float* ad = adst + h * HID;
  float s = 0.f, d = 0.f;
  for (int c = 0; c < HID; ++c) { float v = hp[c]; s += v * as[c]; d += v * ad[c]; }
  es[idx] = s;
  ed[idx] = d;
}

__device__ __forceinline__ void edge_ids(long long i, const int* ei, int& s, int& d) {
  if (i < EE) { s = ei[i]; d = ei[EE + i]; }
  else        { s = (int)(i - EE); d = s; }   /* self loops */
}

__global__ void k_edge_max(const int* __restrict__ ei, const float* __restrict__ es,
                           const float* __restrict__ ed, unsigned* __restrict__ m) {
  long long i = (long long)blockIdx.x * blockDim.x + threadIdx.x;
  if (i >= ET) return;
  int s, d; edge_ids(i, ei, s, d);
  for (int h = 0; h < HEADS; ++h) {
    float e = es[s * 4 + h] + ed[d * 4 + h];
    e = (e > 0.f) ? e : LRELU * e;
    atomicMax(&m[d * 4 + h], fenc(e));
  }
}

__global__ void k_edge_exp(const int* __restrict__ ei, const float* __restrict__ es,
                           const float* __restrict__ ed, const unsigned* __restrict__ m,
                           float* __restrict__ exb, float* __restrict__ denom) {
  long long i = (long long)blockIdx.x * blockDim.x + threadIdx.x;
  if (i >= ET) return;
  int s, d; edge_ids(i, ei, s, d);
  for (int h = 0; h < HEADS; ++h) {
    float e = es[s * 4 + h] + ed[d * 4 + h];
    e = (e > 0.f) ? e : LRELU * e;
    float ex = expf(e - fdec(m[d * 4 + h]));
    exb[i * 4 + h] = ex;
    atomAddF(&denom[d * 4 + h], ex);
  }
}

__global__ void k_edge_agg(const int* __restrict__ ei, const float* __restrict__ exb,
                           const float* __restrict__ denom, const float* __restrict__ H,
                           float* __restrict__ agg) {
  long long j = (long long)blockIdx.x * blockDim.x + threadIdx.x;
  if (j >= (long long)ET * 128) return;
  long long e = j >> 7;
  int c = (int)(j & 127);
  int h = c >> 5;
  int s, d; edge_ids(e, ei, s, d);
  float alpha = exb[e * 4 + h] / denom[d * 4 + h];
  atomAddF(&agg[(size_t)d * 128 + c], alpha * H[(size_t)s * 128 + c]);
}

/* ---------------- bias + div-loss + BN + ELU (+ f16 emit) ---------------- */
__global__ void k_post_gat(const float* __restrict__ agg, const float* __restrict__ bias,
                           const float* __restrict__ proj, float* __restrict__ divacc,
                           const float* __restrict__ gamma, const float* __restrict__ beta,
                           const float* __restrict__ mean, const float* __restrict__ var,
                           float* __restrict__ out_f32, _Float16* __restrict__ out_f16) {
  __shared__ float sdiv[6];
  int t = threadIdx.x;
  if (t < 6) sdiv[t] = 0.f;
  __syncthreads();
  int n = blockIdx.x * blockDim.x + t;
  bool active = (n < NN);
  if (active) {
    const float* ar = agg + (size_t)n * 128;
    /* raw per-head gram (upper triangle incl diag) */
    float D[10];
#pragma unroll
    for (int i = 0; i < 10; ++i) D[i] = 0.f;
    for (int c = 0; c < HID; ++c) {
      float p0 = (ar[c]        + bias[c])        * proj[c];
      float p1 = (ar[32 + c]   + bias[32 + c])   * proj[32 + c];
      float p2 = (ar[64 + c]   + bias[64 + c])   * proj[64 + c];
      float p3 = (ar[96 + c]   + bias[96 + c])   * proj[96 + c];
      D[0] += p0 * p0; D[1] += p0 * p1; D[2] += p0 * p2; D[3] += p0 * p3;
      D[4] += p1 * p1; D[5] += p1 * p2; D[6] += p1 * p3;
      D[7] += p2 * p2; D[8] += p2 * p3; D[9] += p3 * p3;
    }
    float n0 = fmaxf(sqrtf(D[0]), 1e-8f), n1 = fmaxf(sqrtf(D[4]), 1e-8f);
    float n2 = fmaxf(sqrtf(D[7]), 1e-8f), n3 = fmaxf(sqrtf(D[9]), 1e-8f);
    atomAddF(&sdiv[0], D[1] / (n0 * n1));  /* LDS float atomics -> ds_add_f32 */
    atomAddF(&sdiv[1], D[2] / (n0 * n2));
    atomAddF(&sdiv[2], D[3] / (n0 * n3));
    atomAddF(&sdiv[3], D[5] / (n1 * n2));
    atomAddF(&sdiv[4], D[6] / (n1 * n3));
    atomAddF(&sdiv[5], D[8] / (n2 * n3));
  }
  __syncthreads();
  if (t < 6) atomAddF(&divacc[t], sdiv[t]);

  if (!active) {
    if (n < NPAD && out_f16)
      for (int c = 0; c < 128; ++c) out_f16[(size_t)n * 128 + c] = (_Float16)0.f;
    return;
  }
  const float* ar = agg + (size_t)n * 128;
  for (int c = 0; c < 128; ++c) {
    float y  = ar[c] + bias[c];
    float xb = (y - mean[c]) * gamma[c] * rsqrtf(var[c] + BN_EPS) + beta[c];
    float a  = (xb > 0.f) ? xb : (expf(xb) - 1.f);
    out_f32[(size_t)n * 128 + c] = a;
    if (out_f16) out_f16[(size_t)n * 128 + c] = (_Float16)a;
  }
}

/* ---------------- pooling + head ---------------- */
__global__ void k_pool(const float* __restrict__ a, const float* __restrict__ b,
                       const int* __restrict__ batch, float* __restrict__ pool) {
  long long j = (long long)blockIdx.x * blockDim.x + threadIdx.x;
  if (j >= (long long)NN * 128) return;
  int n = (int)(j >> 7);
  int c = (int)(j & 127);
  atomAddF(&pool[(size_t)batch[n] * 128 + c], a[j] + b[j]);
}

__global__ void k_finalize(const float* __restrict__ pool, const int* __restrict__ counts,
                           const float* __restrict__ task_w, const float* __restrict__ task_b,
                           const float* __restrict__ divacc, float* __restrict__ out) {
  __shared__ float se[128];
  int g = blockIdx.x, c = threadIdx.x;
  float cnt = fmaxf((float)counts[g], 1.f);
  float e = pool[(size_t)g * 128 + c] / cnt;
  out[641 + g * 128 + c] = e;      /* emb */
  se[c] = e;
  __syncthreads();
  if (c < OUTD) {
    float t = task_b[c];
    for (int k = 0; k < 128; ++k) t += se[k] * task_w[k * OUTD + c];
    out[g * OUTD + c] = t;         /* task */
  }
  if (g == 0 && c == 0) {
    float d = 0.f;
    for (int i = 0; i < 12; ++i) { float m = divacc[i] / (float)NN; d += m * m; }
    out[640] = 0.1f * d / 6.f;     /* DIV_W * (mean6 + mean6) */
  }
}

/* ================= host side ================= */
static inline int blks(long long n, int b) { return (int)((n + b - 1) / b); }

extern "C" void kernel_launch(void* const* d_in, const int* in_sizes, int n_in,
                              void* d_out, int out_size, void* d_ws, size_t ws_size,
                              hipStream_t stream) {
  (void)in_sizes; (void)n_in; (void)out_size; (void)ws_size;
  const float* x      = (const float*)d_in[0];
  const int*   ei     = (const int*)d_in[1];
  const int*   batch  = (const int*)d_in[2];
  const float* pos_w  = (const float*)d_in[3];
  const float* pos_b  = (const float*)d_in[4];
  const float* w0     = (const float*)d_in[5];
  const float* asrc0  = (const float*)d_in[6];
  const float* adst0  = (const float*)d_in[7];
  const float* b0     = (const float*)d_in[8];
  const float* w1     = (const float*)d_in[9];
  const float* asrc1  = (const float*)d_in[10];
  const float* adst1  = (const float*)d_in[11];
  const float* b1     = (const float*)d_in[12];
  const float* bn_g   = (const float*)d_in[13];
  const float* bn_b   = (const float*)d_in[14];
  const float* bn_m   = (const float*)d_in[15];
  const float* bn_v   = (const float*)d_in[16];
  const float* proj   = (const float*)d_in[17];
  const float* res_w  = (const float*)d_in[18];
  const float* res_b  = (const float*)d_in[19];
  const float* task_w = (const float*)d_in[20];
  const float* task_b = (const float*)d_in[21];
  float* out = (float*)d_out;

  /* workspace carve-out */
  char* base = (char*)d_ws;
  size_t off = 0;
  auto alloc = [&](size_t bytes) -> char* {
    char* p = base + off;
    off = (off + bytes + 255) & ~(size_t)255;
    return p;
  };
  int*      counts = (int*)alloc(GG * 4);
  int*      starts = (int*)alloc(GG * 4);
  _Float16* h0f16  = (_Float16*)alloc((size_t)NPAD * 160 * 2);  /* reused as act f16 (stride 128) */
  float*    H      = (float*)alloc((size_t)NPAD * 128 * 4);
  float*    agg    = (float*)alloc((size_t)NPAD * 128 * 4);
  float*    act    = (float*)alloc((size_t)NPAD * 128 * 4);
  float*    es     = (float*)alloc((size_t)NN * 4 * 4);
  float*    ed     = (float*)alloc((size_t)NN * 4 * 4);
  unsigned* mmax   = (unsigned*)alloc((size_t)NN * 4 * 4);
  float*    denom  = (float*)alloc((size_t)NN * 4 * 4);
  float*    exb    = (float*)alloc((size_t)ET * 4 * 4);
  _Float16* wpack  = (_Float16*)alloc((size_t)5 * 4096 * 2);
  float*    pool   = (float*)alloc((size_t)GG * 128 * 4);
  float*    divacc = (float*)alloc(16 * 4);

  /* ---- init ---- */
  k_fill_u32<<<blks(GG, 64), 64, 0, stream>>>((unsigned*)counts, GG, 0u);
  k_fill_f32<<<blks(GG * 128, 256), 256, 0, stream>>>(pool, GG * 128, 0.f);
  k_fill_f32<<<1, 16, 0, stream>>>(divacc, 12, 0.f);

  /* ---- positions + f16 input ---- */
  k_count<<<blks(NN, 256), 256, 0, stream>>>(batch, counts);
  k_scan<<<1, 64, 0, stream>>>(counts, starts);
  k_build_h0<<<blks(NPAD, 128), 128, 0, stream>>>(x, batch, counts, starts, pos_w, pos_b, h0f16);

  /* ---- layer 0 GEMM: H = h0 @ w0  (K=144 -> 5 K-tiles) ---- */
  k_pack_w<<<blks(5 * 4096, 256), 256, 0, stream>>>(w0, 144, 5, wpack);
  k_gemm_wmma<<<NPAD / 64, 128, 0, stream>>>(h0f16, 160, 5, wpack, nullptr, H);

  /* ---- layer 0 attention ---- */
  k_scores<<<blks((long long)NN * 4, 256), 256, 0, stream>>>(H, asrc0, adst0, es, ed);
  k_fill_f32<<<blks((long long)NPAD * 128, 256), 256, 0, stream>>>(agg, (long long)NPAD * 128, 0.f);
  k_fill_u32<<<blks((long long)NN * 4, 256), 256, 0, stream>>>(mmax, (long long)NN * 4, ENC_NEG_INF);
  k_fill_f32<<<blks((long long)NN * 4, 256), 256, 0, stream>>>(denom, (long long)NN * 4, 0.f);
  k_edge_max<<<blks(ET, 256), 256, 0, stream>>>(ei, es, ed, mmax);
  k_edge_exp<<<blks(ET, 256), 256, 0, stream>>>(ei, es, ed, mmax, exb, denom);
  k_edge_agg<<<blks((long long)ET * 128, 256), 256, 0, stream>>>(ei, exb, denom, H, agg);

  /* ---- layer 0 post: bias, div, BN, ELU -> act(f32) + act(f16) ---- */
  k_post_gat<<<blks(NPAD, 128), 128, 0, stream>>>(agg, b0, proj, divacc,
                                                  bn_g, bn_b, bn_m, bn_v,
                                                  act, h0f16);

  /* ---- layer 1 GEMM: H = act @ w1  (K=128 -> 4 K-tiles) ---- */
  k_pack_w<<<blks(4 * 4096, 256), 256, 0, stream>>>(w1, 128, 4, wpack);
  k_gemm_wmma<<<NPAD / 64, 128, 0, stream>>>(h0f16, 128, 4, wpack, nullptr, H);

  /* ---- layer 1 attention ---- */
  k_scores<<<blks((long long)NN * 4, 256), 256, 0, stream>>>(H, asrc1, adst1, es, ed);
  k_fill_f32<<<blks((long long)NPAD * 128, 256), 256, 0, stream>>>(agg, (long long)NPAD * 128, 0.f);
  k_fill_u32<<<blks((long long)NN * 4, 256), 256, 0, stream>>>(mmax, (long long)NN * 4, ENC_NEG_INF);
  k_fill_f32<<<blks((long long)NN * 4, 256), 256, 0, stream>>>(denom, (long long)NN * 4, 0.f);
  k_edge_max<<<blks(ET, 256), 256, 0, stream>>>(ei, es, ed, mmax);
  k_edge_exp<<<blks(ET, 256), 256, 0, stream>>>(ei, es, ed, mmax, exb, denom);
  k_edge_agg<<<blks((long long)ET * 128, 256), 256, 0, stream>>>(ei, exb, denom, H, agg);

  /* ---- layer 1 post (in-place into agg; keep act f16 for residual GEMM) ---- */
  k_post_gat<<<blks(NPAD, 128), 128, 0, stream>>>(agg, b1, proj + 128, divacc + 6,
                                                  bn_g + 128, bn_b + 128, bn_m + 128, bn_v + 128,
                                                  agg, nullptr);

  /* ---- residual GEMM: H = act(prev,f16) @ res_w + res_b ---- */
  k_pack_w<<<blks(4 * 4096, 256), 256, 0, stream>>>(res_w, 128, 4, wpack);
  k_gemm_wmma<<<NPAD / 64, 128, 0, stream>>>(h0f16, 128, 4, wpack, res_b, H);

  /* ---- pool + head ---- */
  k_pool<<<blks((long long)NN * 128, 256), 256, 0, stream>>>(agg, H, batch, pool);
  k_finalize<<<GG, 128, 0, stream>>>(pool, counts, task_w, task_b, divacc, out);
}
